// EinetMixture_55344948576523
// MI455X (gfx1250) — matrix operand order repaired and checked by reading, not compiled, not run on hardware
//
#include <hip/hip_runtime.h>
#include <hip/hip_bf16.h>
#include <math.h>

// EinetMixture for MI455X (gfx1250): route -> sort by cluster -> per-expert
// bf16 WMMA GEMM (only the assigned expert's 64 leaves per row) -> logsumexp.
//
// GEMM kernel: 128 threads = 4 waves per block; the block stages one 16-row
// A tile ([x^2|x] in bf16) in LDS, each wave computes one 16x16 N-tile with
// v_wmma_f32_16x16x32_bf16 using a single in-place 8-VGPR accumulator.

#define BATCH   32768
#define DFEAT   784
#define NEXP    8
#define NLEAF   64
#define KDIM    1568            // 2 * DFEAT : [x^2 | x]
#define SMSTRIDE 1576           // KDIM + 8 ushorts -> row stride 3152 B (16B aligned)
#define LOG2PI  1.8378770664093453f

typedef __attribute__((ext_vector_type(4)))  unsigned int u32x4;
typedef __attribute__((ext_vector_type(16))) __bf16       v16bf;
typedef __attribute__((ext_vector_type(8)))  float        v8f;

union Frag { u32x4 q[2]; v16bf v; };

__device__ __forceinline__ unsigned short f2bf(float f) {
  union { float f; unsigned u; } v; v.f = f;
  unsigned r = v.u + 0x7FFFu + ((v.u >> 16) & 1u);   // round-to-nearest-even
  return (unsigned short)(r >> 16);
}

// ---------------- small prep kernels ----------------

__global__ void __launch_bounds__(32)
init_kernel(int* counts, int* cursor) {
  int t = threadIdx.x;
  if (t < NEXP) { counts[t] = 0; cursor[t] = 0; }
}

__global__ void __launch_bounds__(256)
cnorm_kernel(const float* __restrict__ cents, float* __restrict__ cnorm) {
  const int e = threadIdx.x >> 5, lane = threadIdx.x & 31;
  float a = 0.f;
  for (int d = lane; d < DFEAT; d += 32) { float c = cents[e * DFEAT + d]; a += c * c; }
  for (int off = 16; off; off >>= 1) a += __shfl_xor(a, off);
  if (lane == 0) cnorm[e] = a;
}

// Wt[e][leaf][0..783] = bf16(-0.5*inv_var), Wt[e][leaf][784..1567] = bf16(mu*inv_var)
__global__ void __launch_bounds__(256)
prepw_kernel(const float* __restrict__ means, const float* __restrict__ log_stds,
             unsigned short* __restrict__ Wt, float* __restrict__ ctmp) {
  const int ek = blockIdx.x;                 // e*NLEAF + leaf
  const int tid = threadIdx.x;
  const size_t base = (size_t)ek * DFEAT;
  float csum = 0.f;
  for (int d = tid; d < DFEAT; d += 256) {
    float ls = log_stds[base + d];
    float mu = means[base + d];
    float iv = __expf(-2.f * ls);
    Wt[(size_t)ek * KDIM + d]         = f2bf(-0.5f * iv);
    Wt[(size_t)ek * KDIM + DFEAT + d] = f2bf(mu * iv);
    csum += -0.5f * mu * mu * iv - ls;
  }
  for (int off = 16; off; off >>= 1) csum += __shfl_xor(csum, off);
  __shared__ float wsum[8];
  if ((tid & 31) == 0) wsum[tid >> 5] = csum;
  __syncthreads();
  if (tid == 0) {
    float t = 0.f;
    for (int w = 0; w < 8; ++w) t += wsum[w];
    ctmp[ek] = t - 0.5f * (float)DFEAT * LOG2PI;
  }
}

// cvec[e][k] = c[e][k] + log_softmax(logits[e])[k]
__global__ void __launch_bounds__(32)
logw_kernel(const float* __restrict__ logits, const float* __restrict__ ctmp,
            float* __restrict__ cvec) {
  const int e = blockIdx.x, t = threadIdx.x;
  float l0 = logits[e * NLEAF + t], l1 = logits[e * NLEAF + 32 + t];
  float m = fmaxf(l0, l1);
  for (int off = 16; off; off >>= 1) m = fmaxf(m, __shfl_xor(m, off));
  float s = __expf(l0 - m) + __expf(l1 - m);
  for (int off = 16; off; off >>= 1) s += __shfl_xor(s, off);
  float lse = m + __logf(s);
  cvec[e * NLEAF + t]      = ctmp[e * NLEAF + t]      + l0 - lse;
  cvec[e * NLEAF + 32 + t] = ctmp[e * NLEAF + 32 + t] + l1 - lse;
}

// ---------------- routing: one wave per row ----------------

__global__ void __launch_bounds__(256)
route_kernel(const float* __restrict__ x, const float* __restrict__ cents,
             const float* __restrict__ cnorm, int* __restrict__ cluster,
             int* __restrict__ counts) {
  __shared__ float sc[NEXP * DFEAT];         // 25 KB
  const int tid = threadIdx.x;
  for (int i = tid; i < NEXP * DFEAT; i += 256) sc[i] = cents[i];
  __syncthreads();
  const int lane = tid & 31;
  const int b = blockIdx.x * 8 + (tid >> 5);
  const float* xr = x + (size_t)b * DFEAT;
  float acc[NEXP];
#pragma unroll
  for (int e = 0; e < NEXP; ++e) acc[e] = 0.f;
  for (int d = lane; d < DFEAT; d += 32) {
    float xv = xr[d];
#pragma unroll
    for (int e = 0; e < NEXP; ++e) acc[e] += xv * sc[e * DFEAT + d];
  }
#pragma unroll
  for (int e = 0; e < NEXP; ++e)
    for (int off = 16; off; off >>= 1) acc[e] += __shfl_xor(acc[e], off);
  if (lane == 0) {
    int best = 0; float bs = cnorm[0] - 2.f * acc[0];
#pragma unroll
    for (int e = 1; e < NEXP; ++e) {
      float sgn = cnorm[e] - 2.f * acc[e];
      if (sgn < bs) { bs = sgn; best = e; }   // first-min tie-break, like argmin
    }
    cluster[b] = best;
    atomicAdd(&counts[best], 1);
  }
}

__global__ void __launch_bounds__(32)
scan_kernel(const int* __restrict__ counts, int* __restrict__ cursor) {
  if (threadIdx.x == 0) {
    int run = 0;
    for (int e = 0; e < NEXP; ++e) { cursor[e] = run; run += counts[e]; }
  }
}

__global__ void __launch_bounds__(256)
scatter_kernel(const int* __restrict__ cluster, int* __restrict__ cursor,
               int* __restrict__ sorted) {
  int b = blockIdx.x * 256 + threadIdx.x;
  if (b < BATCH) {
    int pos = atomicAdd(&cursor[cluster[b]], 1);
    sorted[pos] = b;
  }
}

// ---------------- main WMMA kernel: 4 waves per 16-row tile ----------------

__global__ void __launch_bounds__(128)
einet_gemm_kernel(const float* __restrict__ x,
                  const unsigned short* __restrict__ Wt,
                  const float* __restrict__ cvec,
                  const int* __restrict__ cluster,
                  const int* __restrict__ sorted,
                  float* __restrict__ out) {
  __shared__ __align__(16) unsigned short smA[16][SMSTRIDE]; // 50,432 B
  __shared__ float ctile[16][NLEAF];                          //  4,096 B
  __shared__ int semask;

  const int tid  = threadIdx.x;          // 0..127
  const int lane = tid & 31;
  const int wav  = tid >> 5;             // wave id == N-tile id (0..3)
  const int mrow = lane & 15;            // A-row / B-column index in tile
  const int hi   = lane >> 4;            // lane half selects K sub-range
  const int tileBase = blockIdx.x * 16;

  const int rb = sorted[tileBase + mrow];      // this lane's row's batch index
  const int re = cluster[rb];                  // its expert

  // Block-uniform mask of experts present in this (sorted) tile.
  if (tid == 0) semask = 0;
  __syncthreads();
  if (tid < 16) atomicOr(&semask, 1 << re);

  // Stage A rows as bf16 [x^2 | x] into LDS (coalesced f32 loads, 128 threads).
  for (int m = 0; m < 16; ++m) {
    const int b = __shfl(rb, m);               // row m's batch index (per-wave)
    const float* xr = x + (size_t)b * DFEAT;
    for (int d = tid; d < DFEAT; d += 128) {
      float xv = xr[d];
      smA[m][d]         = f2bf(xv * xv);
      smA[m][DFEAT + d] = f2bf(xv);
    }
  }
  __syncthreads();
  const unsigned em = (unsigned)semask;

  // This wave's B column block: column = wav*16 + mrow, K offset hi*8.
  float result = 0.0f;
  for (int e = 0; e < NEXP; ++e) {
    if (!((em >> e) & 1u)) continue;           // uniform across block

    const unsigned short* wrow =
        Wt + ((size_t)e * NLEAF + wav * 16 + mrow) * KDIM + hi * 8;
    __builtin_prefetch(wrow, 0, 0);            // global_prefetch_b8

    v8f acc = {};
    for (int kb = 0; kb < KDIM; kb += 32) {
      // A fragment: 16-bit A 16x32 layout -> two ds_load_b128 per lane.
      Frag fa;
      fa.q[0] = *(const u32x4*)&smA[mrow][kb + hi * 8];        // K = kb+hi8 .. +7
      fa.q[1] = *(const u32x4*)&smA[mrow][kb + 16 + hi * 8];   // K = kb+16+hi8 .. +7
      // B fragment: K = kb+hi8..+7 and kb+16+hi8..+7 of this column -> two global b128 loads.
      Frag fb;
      fb.q[0] = *(const u32x4*)(wrow + kb);
      fb.q[1] = *(const u32x4*)(wrow + kb + 16);
      acc = __builtin_amdgcn_wmma_f32_16x16x32_bf16(
          false, fa.v, false, fb.v, (short)0, acc, false, false);
    }

    // Spill this wave's 16x16 C tile; C layout: VGPR v -> M = v (+8 for hi lanes).
#pragma unroll
    for (int v = 0; v < 8; ++v)
      ctile[v + hi * 8][wav * 16 + mrow] = acc[v];
    __syncthreads();

    if (tid < 16 && re == e) {                 // tid<16 => wave 0, row == tid
      const float* cv = cvec + e * NLEAF;
      float mx = -3.0e38f;
      for (int k = 0; k < NLEAF; ++k) mx = fmaxf(mx, ctile[tid][k] + cv[k]);
      float s = 0.f;
      for (int k = 0; k < NLEAF; ++k) s += __expf(ctile[tid][k] + cv[k] - mx);
      result = mx + __logf(s);
    }
    __syncthreads();   // ctile reused by next expert pass (boundary tiles only)
  }

  if (tid < 16) out[rb] = result;   // scattered back to original order
}

// ---------------- launch ----------------

extern "C" void kernel_launch(void* const* d_in, const int* in_sizes, int n_in,
                              void* d_out, int out_size, void* d_ws, size_t ws_size,
                              hipStream_t stream) {
  (void)in_sizes; (void)n_in; (void)out_size; (void)ws_size;
  const float* x        = (const float*)d_in[0];
  const float* cents    = (const float*)d_in[1];
  const float* means    = (const float*)d_in[2];
  const float* log_stds = (const float*)d_in[3];
  const float* logits   = (const float*)d_in[4];
  float* out = (float*)d_out;

  char* p = (char*)d_ws;
  auto carve = [&](size_t bytes) -> void* {
    void* r = (void*)p; p += (bytes + 255) & ~(size_t)255; return r;
  };
  unsigned short* Wt = (unsigned short*)carve(sizeof(unsigned short) * NEXP * NLEAF * KDIM);
  float* ctmp   = (float*)carve(sizeof(float) * NEXP * NLEAF);
  float* cvec   = (float*)carve(sizeof(float) * NEXP * NLEAF);
  float* cnorm  = (float*)carve(sizeof(float) * NEXP);
  int*   clustr = (int*)carve(sizeof(int) * BATCH);
  int*   sorted = (int*)carve(sizeof(int) * BATCH);
  int*   counts = (int*)carve(sizeof(int) * NEXP);
  int*   cursor = (int*)carve(sizeof(int) * NEXP);

  init_kernel   <<<1, 32, 0, stream>>>(counts, cursor);
  cnorm_kernel  <<<1, 256, 0, stream>>>(cents, cnorm);
  prepw_kernel  <<<NEXP * NLEAF, 256, 0, stream>>>(means, log_stds, Wt, ctmp);
  logw_kernel   <<<NEXP, 32, 0, stream>>>(logits, ctmp, cvec);
  route_kernel  <<<BATCH / 8, 256, 0, stream>>>(x, cents, cnorm, clustr, counts);
  scan_kernel   <<<1, 32, 0, stream>>>(counts, cursor);
  scatter_kernel<<<BATCH / 256, 256, 0, stream>>>(clustr, cursor, sorted);
  einet_gemm_kernel<<<BATCH / 16, 128, 0, stream>>>(x, Wt, cvec, clustr, sorted, out);
}
